// AttnDecoderRNN_66176856096830
// MI455X (gfx1250) — matrix operand order repaired and checked by reading, compiled, not run on hardware
//
#include <hip/hip_runtime.h>
#include <hip/hip_bf16.h>
#include <math.h>

#define B_ 64
#define S_ 128
#define H_ 1024
#define V_ 32000
#define T_ 10

typedef __bf16 bf16_t;
typedef __attribute__((ext_vector_type(16))) __bf16 v16bf;
typedef __attribute__((ext_vector_type(8)))  __bf16 v8bf;
typedef __attribute__((ext_vector_type(8)))  float  v8f;

// ---------- elementwise helpers ----------

__global__ void f2bf_kernel(const float* __restrict__ src, bf16_t* __restrict__ dst, size_t n) {
  size_t i = (size_t)blockIdx.x * blockDim.x + threadIdx.x;
  if (i < n) dst[i] = (bf16_t)src[i];
}

__global__ void emb_gather_kernel(const float* __restrict__ emb, const int* __restrict__ tgt,
                                  bf16_t* __restrict__ out) {
  int i = blockIdx.x * blockDim.x + threadIdx.x;   // over B*T*H
  if (i >= B_ * T_ * H_) return;
  int h  = i & (H_ - 1);
  int bt = i >> 10;            // b*T + t
  int t  = bt % T_;
  int b  = bt / T_;
  int tok = (t == 0) ? 0 : tgt[b * T_ + (t - 1)];  // SOS = 0
  out[i] = (bf16_t)emb[(size_t)tok * H_ + h];
}

__global__ void init_h_kernel(const float* __restrict__ eh, float* __restrict__ h,
                              bf16_t* __restrict__ hb) {
  int i = blockIdx.x * blockDim.x + threadIdx.x;
  if (i >= B_ * H_) return;
  float v = eh[i];
  h[i] = v;
  hb[i] = (bf16_t)v;
}

__global__ void copy_kernel(const float* __restrict__ s, float* __restrict__ d, size_t n) {
  size_t i = (size_t)blockIdx.x * blockDim.x + threadIdx.x;
  if (i < n) d[i] = s[i];
}

// ---------- generic NT bf16 WMMA GEMM:  C[M,N] = A[M,K] * B[N,K]^T + bias ----------
// Each wave owns a 32(M) x 64(N) tile: 2 M-subtiles x 4 N-subtiles = 8 WMMAs per
// 32-wide K step; A fragments reused 4x, B fragments reused 2x (12 b128 loads / 8 WMMA).

__global__ __launch_bounds__(256)
void wmma_gemm_nt(const bf16_t* __restrict__ A, const bf16_t* __restrict__ Bm,
                  const float* __restrict__ bias, float* __restrict__ C,
                  int M, int N, int K) {
  int wave = (blockIdx.x * blockDim.x + threadIdx.x) >> 5;
  int lane = threadIdx.x & 31;
  int tilesN = N >> 6;   // 64-wide N tiles
  int tilesM = M >> 5;   // 32-tall M tiles
  if (wave >= tilesM * tilesN) return;
  int mt = wave / tilesN;
  int nt = wave - mt * tilesN;

#if defined(__gfx1250__)
  int half = lane >> 4;     // 0: lanes 0-15, 1: lanes 16-31
  int l15  = lane & 15;
  const bf16_t* Arow0 = A  + (size_t)(mt * 32 + l15) * K;        // M rows 0..15
  const bf16_t* Arow1 = A  + (size_t)(mt * 32 + 16 + l15) * K;   // M rows 16..31
  const bf16_t* Bcol  = Bm + (size_t)(nt * 64 + l15) * K;
  int kA = half * 16;       // A layout: lane-half selects K sub-block of 16
  int kB = half * 16;       // B layout: lane-half selects K sub-block of 16

  v8f acc[2][4] = {{v8f{}, v8f{}, v8f{}, v8f{}}, {v8f{}, v8f{}, v8f{}, v8f{}}};

  for (int k0 = 0; k0 < K; k0 += 32) {
    if (k0 + 32 < K) {
      __builtin_prefetch(Arow0 + k0 + 32 + kA, 0, 3);
      __builtin_prefetch(Arow1 + k0 + 32 + kA, 0, 3);
      __builtin_prefetch(Bcol + k0 + 32 + kB, 0, 3);
    }
    // A fragments: 16x32 bf16 tiles. Per-lane: K = {kA..kA+7} and {kA+8..kA+15}
    v8bf a0lo = *(const v8bf*)(Arow0 + k0 + kA);
    v8bf a0hi = *(const v8bf*)(Arow0 + k0 + 8 + kA);
    v8bf a1lo = *(const v8bf*)(Arow1 + k0 + kA);
    v8bf a1hi = *(const v8bf*)(Arow1 + k0 + 8 + kA);
    v16bf a0, a1;
#pragma unroll
    for (int j = 0; j < 8; ++j) {
      a0[j] = a0lo[j]; a0[j + 8] = a0hi[j];
      a1[j] = a1lo[j]; a1[j + 8] = a1hi[j];
    }
    // B fragments: 32x16 tiles, column (nt*64 + s*16 + l15), contiguous K run of 16
    v16bf bfrag[4];
#pragma unroll
    for (int s = 0; s < 4; ++s)
      bfrag[s] = *(const v16bf*)(Bcol + (size_t)(16 * s) * K + k0 + kB);

#pragma unroll
    for (int s = 0; s < 4; ++s) {
      acc[0][s] = __builtin_amdgcn_wmma_f32_16x16x32_bf16(
          false, a0, false, bfrag[s], (short)0, acc[0][s], false, false);
      acc[1][s] = __builtin_amdgcn_wmma_f32_16x16x32_bf16(
          false, a1, false, bfrag[s], (short)0, acc[1][s], false, false);
    }
  }

  // D layout: VGPR r -> M = tile + half*8 + r, N = nt*64 + s*16 + l15
#pragma unroll
  for (int s = 0; s < 4; ++s) {
    int col = nt * 64 + s * 16 + l15;
    float bv = bias ? bias[col] : 0.0f;
#pragma unroll
    for (int m = 0; m < 2; ++m) {
#pragma unroll
      for (int r = 0; r < 8; ++r) {
        C[(size_t)(mt * 32 + m * 16 + half * 8 + r) * N + col] = acc[m][s][r] + bv;
      }
    }
  }
#else
  // Host-pass / non-gfx1250 scalar fallback (never the perf path)
  if (lane == 0) {
    for (int mm = 0; mm < 32; ++mm)
      for (int nn = 0; nn < 64; ++nn) {
        float acc = bias ? bias[nt * 64 + nn] : 0.0f;
        const bf16_t* ar = A  + (size_t)(mt * 32 + mm) * K;
        const bf16_t* br = Bm + (size_t)(nt * 64 + nn) * K;
        for (int k = 0; k < K; ++k) acc += (float)ar[k] * (float)br[k];
        C[(size_t)(mt * 32 + mm) * N + nt * 64 + nn] = acc;
      }
  }
#endif
}

// ---------- attention ----------

__device__ inline float wave_reduce_sum(float v) {
#pragma unroll
  for (int off = 16; off > 0; off >>= 1) v += __shfl_down(v, off, 32);
  return v;
}

// one wave per (b,s):  scores[b,s] = va . tanh(q[b] + keys[b,s]) + Va_b
__global__ __launch_bounds__(256)
void attn_score_kernel(const float* __restrict__ q, const float* __restrict__ keys,
                       const float* __restrict__ va, const float* __restrict__ vab,
                       float* __restrict__ scores) {
  int wave = (blockIdx.x * blockDim.x + threadIdx.x) >> 5;
  int lane = threadIdx.x & 31;
  if (wave >= B_ * S_) return;
  int b = wave >> 7;
  const float* kp = keys + (size_t)wave * H_;
  const float* qp = q + (size_t)b * H_;
  float acc = 0.0f;
  for (int h = lane; h < H_; h += 32) acc += va[h] * tanhf(qp[h] + kp[h]);
  acc = wave_reduce_sum(acc);
  if (lane == 0) scores[wave] = acc + vab[0];
}

// one block (128 threads) per b: softmax over S; write weights + attns output
__global__ void softmax_attn_kernel(const float* __restrict__ scores, float* __restrict__ wts,
                                    float* __restrict__ attn, int t) {
  __shared__ float red[S_];
  int b = blockIdx.x, s = threadIdx.x;
  float v = scores[b * S_ + s];
  red[s] = v;
  __syncthreads();
  for (int off = 64; off > 0; off >>= 1) {
    if (s < off) red[s] = fmaxf(red[s], red[s + off]);
    __syncthreads();
  }
  float mx = red[0];
  __syncthreads();
  float e = expf(v - mx);
  red[s] = e;
  __syncthreads();
  for (int off = 64; off > 0; off >>= 1) {
    if (s < off) red[s] += red[s + off];
    __syncthreads();
  }
  float w = e / red[0];
  wts[b * S_ + s] = w;
  attn[((size_t)b * T_ + t) * S_ + s] = w;  // attns[b,t,s]
}

// ctx[b,h] = sum_s w[b,s] * enc[b,s,h]
__global__ void ctx_kernel(const float* __restrict__ wts, const float* __restrict__ enc,
                           float* __restrict__ ctx) {
  int i = blockIdx.x * blockDim.x + threadIdx.x;  // B*H
  if (i >= B_ * H_) return;
  int b = i >> 10, h = i & 1023;
  const float* ep = enc + (size_t)b * S_ * H_ + h;
  const float* wp = wts + b * S_;
  float acc = 0.0f;
#pragma unroll 4
  for (int s = 0; s < S_; ++s) acc += wp[s] * ep[(size_t)s * H_];
  ctx[i] = acc;
}

// x = concat(emb_t, ctx) in bf16
__global__ void build_x_kernel(const bf16_t* __restrict__ embbf, const float* __restrict__ ctx,
                               bf16_t* __restrict__ x, int t) {
  int i = blockIdx.x * blockDim.x + threadIdx.x;  // B*2H
  if (i >= B_ * 2 * H_) return;
  int b = i >> 11, j = i & 2047;
  x[i] = (j < H_) ? embbf[((size_t)b * T_ + t) * H_ + j]
                  : (bf16_t)ctx[(size_t)b * H_ + (j - H_)];
}

// GRU gates + hidden update; store h (f32 + bf16) and outs row (bf16) for logits GEMM
__global__ void gate_kernel(const float* __restrict__ gi, const float* __restrict__ gh,
                            float* __restrict__ h, bf16_t* __restrict__ hb,
                            bf16_t* __restrict__ outsb, int t) {
  int i = blockIdx.x * blockDim.x + threadIdx.x;  // B*H
  if (i >= B_ * H_) return;
  int b = i >> 10, hh = i & 1023;
  size_t base = (size_t)b * 3 * H_;
  float ir = gi[base + hh], iz = gi[base + H_ + hh], in_ = gi[base + 2 * H_ + hh];
  float hr = gh[base + hh], hz = gh[base + H_ + hh], hn = gh[base + 2 * H_ + hh];
  float r = 1.0f / (1.0f + expf(-(ir + hr)));
  float z = 1.0f / (1.0f + expf(-(iz + hz)));
  float n = tanhf(in_ + r * hn);
  float hnew = (1.0f - z) * n + z * h[i];
  h[i] = hnew;
  hb[i] = (bf16_t)hnew;
  outsb[((size_t)b * T_ + t) * H_ + hh] = (bf16_t)hnew;  // row index = b*T+t
}

// in-place log-softmax over V per row
__global__ __launch_bounds__(256)
void logsoftmax_kernel(float* __restrict__ logits) {
  __shared__ float red[256];
  int row = blockIdx.x, tid = threadIdx.x;
  float* p = logits + (size_t)row * V_;
  float mx = -3.4e38f;
  for (int i = tid; i < V_; i += 256) mx = fmaxf(mx, p[i]);
  red[tid] = mx;
  __syncthreads();
  for (int off = 128; off > 0; off >>= 1) {
    if (tid < off) red[tid] = fmaxf(red[tid], red[tid + off]);
    __syncthreads();
  }
  mx = red[0];
  __syncthreads();
  float sum = 0.0f;
  for (int i = tid; i < V_; i += 256) sum += expf(p[i] - mx);
  red[tid] = sum;
  __syncthreads();
  for (int off = 128; off > 0; off >>= 1) {
    if (tid < off) red[tid] += red[tid + off];
    __syncthreads();
  }
  float lse = mx + logf(red[0]);
  for (int i = tid; i < V_; i += 256) p[i] -= lse;
}

// ---------- host ----------

extern "C" void kernel_launch(void* const* d_in, const int* in_sizes, int n_in,
                              void* d_out, int out_size, void* d_ws, size_t ws_size,
                              hipStream_t stream) {
  const float* enc   = (const float*)d_in[0];
  const float* ehid  = (const float*)d_in[1];
  const int*   tgt   = (const int*)  d_in[2];
  const float* emb   = (const float*)d_in[3];
  const float* Wa_b  = (const float*)d_in[5];
  const float* Ua_b  = (const float*)d_in[7];
  const float* Va_w  = (const float*)d_in[8];
  const float* Va_b  = (const float*)d_in[9];
  const float* g_bi  = (const float*)d_in[12];
  const float* g_bh  = (const float*)d_in[13];
  const float* out_b = (const float*)d_in[15];

  char* ws = (char*)d_ws;
  size_t off = 0;
  auto alloc = [&](size_t bytes) -> void* {
    void* p = ws + off;
    off = (off + bytes + 255) & ~(size_t)255;
    return p;
  };

  bf16_t* enc_bf  = (bf16_t*)alloc((size_t)B_ * S_ * H_ * 2);
  bf16_t* Uaw_bf  = (bf16_t*)alloc((size_t)H_ * H_ * 2);
  bf16_t* Waw_bf  = (bf16_t*)alloc((size_t)H_ * H_ * 2);
  bf16_t* wi_bf   = (bf16_t*)alloc((size_t)3 * H_ * 2 * H_ * 2);
  bf16_t* wh_bf   = (bf16_t*)alloc((size_t)3 * H_ * H_ * 2);
  bf16_t* ow_bf   = (bf16_t*)alloc((size_t)V_ * H_ * 2);
  bf16_t* embt_bf = (bf16_t*)alloc((size_t)B_ * T_ * H_ * 2);
  bf16_t* outs_bf = (bf16_t*)alloc((size_t)B_ * T_ * H_ * 2);
  float*  keys    = (float*) alloc((size_t)B_ * S_ * H_ * 4);
  float*  hbuf    = (float*) alloc((size_t)B_ * H_ * 4);
  bf16_t* h_bf    = (bf16_t*)alloc((size_t)B_ * H_ * 2);
  bf16_t* x_bf    = (bf16_t*)alloc((size_t)B_ * 2 * H_ * 2);
  float*  qbuf    = (float*) alloc((size_t)B_ * H_ * 4);
  float*  scores  = (float*) alloc((size_t)B_ * S_ * 4);
  float*  wts     = (float*) alloc((size_t)B_ * S_ * 4);
  float*  ctxb    = (float*) alloc((size_t)B_ * H_ * 4);
  float*  gi      = (float*) alloc((size_t)B_ * 3 * H_ * 4);
  float*  gh      = (float*) alloc((size_t)B_ * 3 * H_ * 4);

  float* logits = (float*)d_out;                       // [B,T,V]
  float* hfin   = logits + (size_t)B_ * T_ * V_;       // [1,B,H]
  float* attn   = hfin + (size_t)B_ * H_;              // [B,T,S]

  auto cvt = [&](const float* s, bf16_t* d, size_t n) {
    f2bf_kernel<<<(unsigned)((n + 255) / 256), 256, 0, stream>>>(s, d, n);
  };
  cvt(enc,                        enc_bf, (size_t)B_ * S_ * H_);
  cvt((const float*)d_in[6],      Uaw_bf, (size_t)H_ * H_);
  cvt((const float*)d_in[4],      Waw_bf, (size_t)H_ * H_);
  cvt((const float*)d_in[10],     wi_bf,  (size_t)3 * H_ * 2 * H_);
  cvt((const float*)d_in[11],     wh_bf,  (size_t)3 * H_ * H_);
  cvt((const float*)d_in[14],     ow_bf,  (size_t)V_ * H_);

  emb_gather_kernel<<<(B_ * T_ * H_) / 256, 256, 0, stream>>>(emb, tgt, embt_bf);
  init_h_kernel<<<(B_ * H_) / 256, 256, 0, stream>>>(ehid, hbuf, h_bf);

  auto gemm = [&](const bf16_t* A, const bf16_t* Bm, const float* bias, float* C,
                  int M, int N, int K) {
    int waves = (M >> 5) * (N >> 6);
    wmma_gemm_nt<<<(waves + 7) / 8, 256, 0, stream>>>(A, Bm, bias, C, M, N, K);
  };

  // Ua_keys = enc @ Ua_w^T + Ua_b   (M=8192, N=1024, K=1024)
  gemm(enc_bf, Uaw_bf, Ua_b, keys, B_ * S_, H_, H_);

  for (int t = 0; t < T_; ++t) {
    gemm(h_bf, Waw_bf, Wa_b, qbuf, B_, H_, H_);                               // q
    attn_score_kernel<<<(B_ * S_) / 8, 256, 0, stream>>>(qbuf, keys, Va_w, Va_b, scores);
    softmax_attn_kernel<<<B_, S_, 0, stream>>>(scores, wts, attn, t);
    ctx_kernel<<<(B_ * H_) / 256, 256, 0, stream>>>(wts, enc, ctxb);
    build_x_kernel<<<(B_ * 2 * H_) / 256, 256, 0, stream>>>(embt_bf, ctxb, x_bf, t);
    gemm(x_bf, wi_bf, g_bi, gi, B_, 3 * H_, 2 * H_);                          // gi
    gemm(h_bf, wh_bf, g_bh, gh, B_, 3 * H_, H_);                              // gh
    gate_kernel<<<(B_ * H_) / 256, 256, 0, stream>>>(gi, gh, hbuf, h_bf, outs_bf, t);
  }

  // logits = outs @ out_w^T + out_b   (M=640, N=32000, K=1024), straight into d_out
  gemm(outs_bf, ow_bf, out_b, logits, B_ * T_, V_, H_);
  logsoftmax_kernel<<<B_ * T_, 256, 0, stream>>>(logits);
  copy_kernel<<<(B_ * H_) / 256, 256, 0, stream>>>(hbuf, hfin, (size_t)B_ * H_);
}
